// _MLP_6150393168108
// MI455X (gfx1250) — compile-verified, matrix operand order checked by reading, not run on hardware
//
#include <hip/hip_runtime.h>
#include <hip/hip_bf16.h>
#include <math.h>

typedef __attribute__((ext_vector_type(16))) _Float16 v16h;
typedef __attribute__((ext_vector_type(8)))  _Float16 v8h;
typedef __attribute__((ext_vector_type(8)))  float    v8f;
typedef __attribute__((ext_vector_type(4)))  int      v4i;

#define BM 128
#define BN 128
#define BK 64
#define LDSK 72          // BK + 8 f16 pad: row stride 144B (16B-aligned), bank-spread
#define THREADS 256

#if __has_builtin(__builtin_amdgcn_global_load_async_to_lds_b128) && \
    __has_builtin(__builtin_amdgcn_s_wait_asynccnt)
#define ASYNC_LDS 1
__device__ __forceinline__ void async_cp16(const _Float16* g, _Float16* l) {
    void* gv = (void*)const_cast<_Float16*>(g);
    __builtin_amdgcn_global_load_async_to_lds_b128(
        (__attribute__((address_space(1))) v4i*)gv,
        (__attribute__((address_space(3))) v4i*)l, 0, 0);
}
__device__ __forceinline__ void async_wait_le8() { __builtin_amdgcn_s_wait_asynccnt(8); }
__device__ __forceinline__ void async_wait_0()   { __builtin_amdgcn_s_wait_asynccnt(0); }
#else
#define ASYNC_LDS 0
#endif

__device__ __forceinline__ float gelu_erf(float v) {
    return 0.5f * v * (1.0f + erff(v * 0.70710678118654752440f));
}

// ---------------------------------------------------------------------------
// Prepass converters (run once; amortized over all tile re-reads).
// int8 magnitudes (|w|<=127) are exactly representable in f16.
// ---------------------------------------------------------------------------
__global__ __launch_bounds__(256)
void cvt_f32_to_f16(const float* __restrict__ s, _Float16* __restrict__ d, long long n) {
    long long i = ((long long)blockIdx.x * 256 + threadIdx.x) * 8;
    if (i + 8 > n) return;
    const float4 a = *(const float4*)(s + i);
    const float4 b = *(const float4*)(s + i + 4);
    v8h o;
    o[0] = (_Float16)a.x; o[1] = (_Float16)a.y; o[2] = (_Float16)a.z; o[3] = (_Float16)a.w;
    o[4] = (_Float16)b.x; o[5] = (_Float16)b.y; o[6] = (_Float16)b.z; o[7] = (_Float16)b.w;
    *(v8h*)(d + i) = o;
}

__global__ __launch_bounds__(256)
void cvt_i32_to_f16(const int* __restrict__ s, _Float16* __restrict__ d, long long n) {
    long long i = ((long long)blockIdx.x * 256 + threadIdx.x) * 8;
    if (i + 8 > n) return;
    const int4 a = *(const int4*)(s + i);
    const int4 b = *(const int4*)(s + i + 4);
    v8h o;
    o[0] = (_Float16)(float)a.x; o[1] = (_Float16)(float)a.y;
    o[2] = (_Float16)(float)a.z; o[3] = (_Float16)(float)a.w;
    o[4] = (_Float16)(float)b.x; o[5] = (_Float16)(float)b.y;
    o[6] = (_Float16)(float)b.z; o[7] = (_Float16)(float)b.w;
    *(v8h*)(d + i) = o;
}

// ---------------------------------------------------------------------------
// Tiled WMMA GEMM:  Out = epilogue( A[M,K] @ B[N,K]^T * scale[n] + bias[n] )
// f16xf16 path: double-buffered LDS fed by GLOBAL_LOAD_ASYNC_TO_LDS_B128
// (ASYNCcnt pipeline).  Converting paths: register-staged global->LDS pipeline.
// 256 threads = 8 waves (4x2); each wave owns a 32x64 tile = 2x4 WMMA accums.
// ---------------------------------------------------------------------------
template <typename AT, typename BT, bool DOGELU, typename OT>
__global__ __launch_bounds__(THREADS)
void gemm_wmma(const AT* __restrict__ A, const BT* __restrict__ B,
               const float* __restrict__ scale, const float* __restrict__ bias,
               OT* __restrict__ Out, int M, int K, int N)
{
#if ASYNC_LDS
    constexpr bool USE_ASYNC = __is_same(AT, _Float16) && __is_same(BT, _Float16);
#else
    constexpr bool USE_ASYNC = false;
#endif
    constexpr int NBUF = USE_ASYNC ? 2 : 1;

    __shared__ __align__(16) _Float16 As[NBUF][BM][LDSK];
    __shared__ __align__(16) _Float16 Bs[NBUF][BN][LDSK];

    const int tid  = threadIdx.x;
    const int lane = tid & 31;
    const int wid  = tid >> 5;       // 0..7
    const int wm   = wid & 3;        // 4 wave-rows -> 32 M each
    const int wn   = wid >> 2;       // 2 wave-cols -> 64 N each
    const int half = lane >> 4;      // K-half select (ISA 16-bit frag layout)
    const int lrow = lane & 15;

    const int m0 = blockIdx.y * BM;
    const int n0 = blockIdx.x * BN;
    const int T  = K / BK;

    v8f acc[2][4];
    #pragma unroll
    for (int i = 0; i < 2; ++i)
        #pragma unroll
        for (int j = 0; j < 4; ++j)
            #pragma unroll
            for (int r = 0; r < 8; ++r) acc[i][j][r] = 0.0f;

    // ---- compute one staged K-tile: 2 K-slices of 32 -> 16 WMMAs per wave ----
    auto compute_tile = [&](const _Float16 (*as)[LDSK], const _Float16 (*bs)[LDSK]) {
        #pragma unroll
        for (int ks = 0; ks < BK; ks += 32) {
            // lanes 0-15: K 0-7 & 16-23; lanes 16-31: K 8-15 & 24-31
            const int c0 = ks + half * 8;
            v16h afrag[2];
            #pragma unroll
            for (int i = 0; i < 2; ++i) {
                const int row = wm * 32 + i * 16 + lrow;
                ((float4*)&afrag[i])[0] = *(const float4*)&as[row][c0];
                ((float4*)&afrag[i])[1] = *(const float4*)&as[row][c0 + 16];
            }
            v16h bfrag[4];
            #pragma unroll
            for (int j = 0; j < 4; ++j) {
                const int row = wn * 64 + j * 16 + lrow;
                ((float4*)&bfrag[j])[0] = *(const float4*)&bs[row][c0];
                ((float4*)&bfrag[j])[1] = *(const float4*)&bs[row][c0 + 16];
            }
            #pragma unroll
            for (int i = 0; i < 2; ++i)
                #pragma unroll
                for (int j = 0; j < 4; ++j)
                    acc[i][j] = __builtin_amdgcn_wmma_f32_16x16x32_f16(
                        false, afrag[i], false, bfrag[j], (short)0, acc[i][j], false, false);
        }
    };

#if ASYNC_LDS
    if constexpr (USE_ASYNC) {
        // ---- async double-buffered pipeline: 8 ASYNCcnt ops per stage ----
        auto stage = [&](int buf, int k0) {
            const int r = tid >> 3;            // 8 thr/row * 8 f16 (16B)
            const int c = (tid & 7) << 3;
            #pragma unroll
            for (int i = 0; i < 4; ++i) {
                const int row = r + i * 32;
                async_cp16(A + (size_t)(m0 + row) * K + (k0 + c), &As[buf][row][c]);
            }
            #pragma unroll
            for (int i = 0; i < 4; ++i) {
                const int row = r + i * 32;
                async_cp16(B + (size_t)(n0 + row) * K + (k0 + c), &Bs[buf][row][c]);
            }
        };
        stage(0, 0);
        for (int kt = 0; kt < T; ++kt) {
            const int c = kt & 1;
            if (kt + 1 < T) stage(c ^ 1, (kt + 1) * BK);     // overlap next tile
            if (kt + 1 < T) async_wait_le8(); else async_wait_0();  // tile kt landed
            __syncthreads();                                  // all waves staged
            compute_tile(As[c], Bs[c]);
            __syncthreads();                                  // all waves consumed
        }
    } else
#endif
    {
        // ---- register-staged pipeline (converting loaders), single buffer ----
        constexpr int ANV = __is_same(AT, float) ? 8 : 4;
        float4 aR[ANV];
        int4   bRi[8];
        float4 bRf[4];

        auto load_regs = [&](int k0) {
            if constexpr (__is_same(AT, float)) {
                const int r = tid >> 4, c = (tid & 15) << 2;
                #pragma unroll
                for (int i = 0; i < 8; ++i)
                    aR[i] = *(const float4*)(A + (size_t)(m0 + r + i * 16) * K + (k0 + c));
            } else {
                const int r = tid >> 3, c = (tid & 7) << 3;
                #pragma unroll
                for (int i = 0; i < 4; ++i)
                    aR[i] = *(const float4*)(A + (size_t)(m0 + r + i * 32) * K + (k0 + c));
            }
            if constexpr (__is_same(BT, int)) {
                const int r = tid >> 4, c = (tid & 15) << 2;
                #pragma unroll
                for (int i = 0; i < 8; ++i)
                    bRi[i] = *(const int4*)(B + (size_t)(n0 + r + i * 16) * K + (k0 + c));
            } else {
                const int r = tid >> 3, c = (tid & 7) << 3;
                #pragma unroll
                for (int i = 0; i < 4; ++i)
                    bRf[i] = *(const float4*)(B + (size_t)(n0 + r + i * 32) * K + (k0 + c));
            }
        };
        auto commit_regs = [&]() {
            if constexpr (__is_same(AT, float)) {
                const int r = tid >> 4, c = (tid & 15) << 2;
                #pragma unroll
                for (int i = 0; i < 8; ++i) {
                    const int row = r + i * 16;
                    As[0][row][c + 0] = (_Float16)aR[i].x;
                    As[0][row][c + 1] = (_Float16)aR[i].y;
                    As[0][row][c + 2] = (_Float16)aR[i].z;
                    As[0][row][c + 3] = (_Float16)aR[i].w;
                }
            } else {
                const int r = tid >> 3, c = (tid & 7) << 3;
                #pragma unroll
                for (int i = 0; i < 4; ++i)
                    *(float4*)&As[0][r + i * 32][c] = aR[i];
            }
            if constexpr (__is_same(BT, int)) {
                const int r = tid >> 4, c = (tid & 15) << 2;
                #pragma unroll
                for (int i = 0; i < 8; ++i) {
                    const int row = r + i * 16;
                    Bs[0][row][c + 0] = (_Float16)(float)bRi[i].x;
                    Bs[0][row][c + 1] = (_Float16)(float)bRi[i].y;
                    Bs[0][row][c + 2] = (_Float16)(float)bRi[i].z;
                    Bs[0][row][c + 3] = (_Float16)(float)bRi[i].w;
                }
            } else {
                const int r = tid >> 3, c = (tid & 7) << 3;
                #pragma unroll
                for (int i = 0; i < 4; ++i)
                    *(float4*)&Bs[0][r + i * 32][c] = bRf[i];
            }
        };

        load_regs(0);
        for (int kt = 0; kt < T; ++kt) {
            commit_regs();                          // prev consumers done (barrier)
            if (kt + 1 < T) load_regs((kt + 1) * BK);  // overlap with compute
            __syncthreads();
            compute_tile(As[0], Bs[0]);
            __syncthreads();
        }
    }

    // ---- epilogue: per-out-channel dequant scale + bias (+ GELU) ----
    // C/D layout: lane = col, VGPR r -> row r + 8*(lane>=16).
    #pragma unroll
    for (int i = 0; i < 2; ++i) {
        #pragma unroll
        for (int j = 0; j < 4; ++j) {
            const int gcol = n0 + wn * 64 + j * 16 + lrow;
            const float sc = scale[gcol];
            const float bi = bias[gcol];
            const int rbase = m0 + wm * 32 + i * 16 + half * 8;
            #pragma unroll
            for (int r = 0; r < 8; ++r) {
                const float v = acc[i][j][r] * sc + bi;
                if constexpr (DOGELU) {
                    Out[(size_t)(rbase + r) * N + gcol] = (OT)gelu_erf(v);
                } else {
                    Out[(size_t)(rbase + r) * N + gcol] = (OT)v;
                }
            }
        }
    }
}

extern "C" void kernel_launch(void* const* d_in, const int* in_sizes, int n_in,
                              void* d_out, int out_size, void* d_ws, size_t ws_size,
                              hipStream_t stream) {
    const float* x  = (const float*)d_in[0];
    const int*   w1 = (const int*)  d_in[1];
    const float* s1 = (const float*)d_in[2];
    const float* b1 = (const float*)d_in[3];
    const int*   w2 = (const int*)  d_in[4];
    const float* s2 = (const float*)d_in[5];
    const float* b2 = (const float*)d_in[6];
    float*       y  = (float*)d_out;

    const int H = in_sizes[2];           // 4096
    const int D = in_sizes[5];           // 1024
    const int M = in_sizes[0] / D;       // 16384

    const long long nX = (long long)M * D;
    const long long nW = (long long)H * D;   // same for w1 and w2
    const long long nH = (long long)M * H;

    char* ws = (char*)d_ws;
    _Float16* h = (_Float16*)ws;                           // [M,H] f16, 134 MB
    const size_t needPre = (size_t)nH * 2 + (size_t)nX * 2 + (size_t)nW * 4;

    const dim3 grid1(H / BN, M / BM);
    const dim3 grid2(D / BN, M / BM);

    if (ws_size >= needPre) {
        // ---- preconverted f16 x f16 path: async-staged, VALU-free inner loops ----
        _Float16* x16 = (_Float16*)(ws + (size_t)nH * 2);
        _Float16* w1h = (_Float16*)(ws + (size_t)nH * 2 + (size_t)nX * 2);
        _Float16* w2h = (_Float16*)(ws + (size_t)nH * 2 + (size_t)nX * 2 + (size_t)nW * 2);

        cvt_f32_to_f16<<<(int)((nX / 8 + 255) / 256), 256, 0, stream>>>(x, x16, nX);
        cvt_i32_to_f16<<<(int)((nW / 8 + 255) / 256), 256, 0, stream>>>(w1, w1h, nW);
        cvt_i32_to_f16<<<(int)((nW / 8 + 255) / 256), 256, 0, stream>>>(w2, w2h, nW);

        gemm_wmma<_Float16, _Float16, true, _Float16>
            <<<grid1, THREADS, 0, stream>>>(x16, w1h, s1, b1, h, M, D, H);
        gemm_wmma<_Float16, _Float16, false, float>
            <<<grid2, THREADS, 0, stream>>>(h, w2h, s2, b2, y, M, H, D);
    } else {
        // ---- fallback: convert inside the loaders (workspace only holds h) ----
        gemm_wmma<float, int, true, _Float16>
            <<<grid1, THREADS, 0, stream>>>(x, w1, s1, b1, h, M, D, H);
        gemm_wmma<_Float16, int, false, float>
            <<<grid2, THREADS, 0, stream>>>(h, w2, s2, b2, y, M, H, D);
    }
}
